// Chapter4WindPowerModel_34024730919674
// MI455X (gfx1250) — matrix operand order repaired and checked by reading, not compiled, not run on hardware
//
#include <hip/hip_runtime.h>
#include <hip/hip_bf16.h>
#include <math.h>

// ---------------------------------------------------------------------------
// Chapter4WindPowerModel forward, CDNA5 (gfx1250) WMMA implementation.
//
// Pipeline (all on `stream`, serialized by stream order):
//   1) decompose_kernel : x[B,14,N,T] -> hf/daily/trend as f16, [sig][b*n][c][t]
//   2) repack_w1/w2     : conv weights -> per-WMMA-fragment f16 layout
//   3) encoder_kernel   : fused conv1(k=3,GELU)+conv2(k=3,GELU) per 64-wide
//                         T chunk, entirely through LDS; emits last-12 slice
//                         + per-chunk channel-sum partials. All matrix math
//                         is v_wmma_f32_16x16x32_f16.
//   4) reduce_mean      : deterministic fixed-order partial reduction
//   5) head_kernel      : predictor heads + gate softmax, only last 12 t
//   6) proj_kernel      : 384->384 GELU ->64 MLP + L2 normalize
//
// GELU is the branch-free tanh form via native v_exp_f32 + v_rcp_f32 (the
// libm erff path is EXEC-divergent and ~40 VALU ops/element, which would
// out-cost the WMMA pipe on the conv epilogues).
// ---------------------------------------------------------------------------

typedef _Float16 v16h __attribute__((ext_vector_type(16)));
typedef _Float16 v8h  __attribute__((ext_vector_type(8)));
typedef float    v8f  __attribute__((ext_vector_type(8)));

#define B_    8
#define CIN_  14
#define N_    100
#define T_    576
#define HID_  128
#define HZ_   12
#define PER_  144
#define BN_   800          // B_*N_
#define NCH_  9            // T_/64 chunks
#define LAST0 (T_ - HZ_)   // 564

__device__ __forceinline__ float gelu_f(float x) {
  // gelu(x) = x * sigmoid(2*0.7978845608*(x + 0.044715 x^3))
  // branch-free: one v_exp_f32 + one v_rcp_f32, no EXEC divergence.
  const float u = 1.5957691216057308f * (x + 0.044715f * x * x * x);
  const float e = __expf(-u);
  return x * __builtin_amdgcn_rcpf(1.0f + e);
}

__device__ __forceinline__ v8f wmma_f16(v16h a, v16h b, v8f c) {
  // D = A(16x32 f16) * B(32x16 f16) + C(16x16 f32)
  return __builtin_amdgcn_wmma_f32_16x16x32_f16(
      /*neg_a=*/false, a, /*neg_b=*/false, b,
      /*c_mod=*/(short)0, c, /*reuse_a=*/false, /*reuse_b=*/false);
}

struct W3      { const float* p[3]; };
struct EncArgs { const float* b1[3]; const float* b2[3]; };
struct HeadArgs {
  const float *gw1, *gb1, *gw2, *gb2;
  const float *pw1[3], *pb1[3], *pw2[3], *pb2[3];
};

// ---------------------------------------------------------------------------
// 1) decompose: trend (avg-25, count_include_pad), daily (per-phase mean over
//    4 full periods), hf = x - 0.5*(trend+daily). Output f16.
// ---------------------------------------------------------------------------
__global__ __launch_bounds__(T_)
void decompose_kernel(const float* __restrict__ x, _Float16* __restrict__ sig) {
  const int c = blockIdx.x, n = blockIdx.y, b = blockIdx.z;
  const int t = threadIdx.x;
  __shared__ float row[T_];
  row[t] = x[(((size_t)b * CIN_ + c) * N_ + n) * T_ + t];
  __syncthreads();

  float tr = 0.f;
#pragma unroll
  for (int j = -12; j <= 12; ++j) {
    int u = t + j;
    if (u >= 0 && u < T_) tr += row[u];
  }
  tr *= (1.0f / 25.0f);

  const int ph = t % PER_;
  const float dl = 0.25f * (row[ph] + row[ph + PER_] + row[ph + 2 * PER_] + row[ph + 3 * PER_]);
  const float hf = row[t] - 0.5f * (tr + dl);

  const size_t base = (((size_t)(b * N_ + n)) * CIN_ + c) * T_ + t;
  const size_t stride = (size_t)BN_ * CIN_ * T_;
  sig[base]              = (_Float16)hf;
  sig[stride + base]     = (_Float16)dl;
  sig[2 * stride + base] = (_Float16)tr;
}

// ---------------------------------------------------------------------------
// 2) Weight repack into WMMA A-fragment order.
// A-frag element(lane,h): m = mtile*16 + lane%16
//                         k = 32*ks + (lane/16)*16 + h
// conv1: k = dt*14 + c (K=42 padded to 64); conv2: k = dt*128 + c (K=384).
// ---------------------------------------------------------------------------
__global__ void repack_w1_kernel(W3 w, _Float16* __restrict__ wrep1) {
  int idx = blockIdx.x * 256 + threadIdx.x;       // 3*8*2*512 = 24576
  if (idx >= 3 * 8 * 2 * 512) return;
  const int h = idx & 15, lane = (idx >> 4) & 31, frag = idx >> 9;
  const int ks = frag & 1, mtile = (frag >> 1) & 7, sg = frag >> 4;
  const int m = mtile * 16 + (lane & 15);
  const int k = 32 * ks + (lane >> 4) * 16 + h;
  float v = 0.f;
  if (k < 42) { int dt = k / CIN_, c = k % CIN_; v = w.p[sg][(m * CIN_ + c) * 3 + dt]; }
  wrep1[idx] = (_Float16)v;
}

__global__ void repack_w2_kernel(W3 w, _Float16* __restrict__ wrep2) {
  int idx = blockIdx.x * 256 + threadIdx.x;       // 3*8*12*512 = 147456
  if (idx >= 3 * 8 * 12 * 512) return;
  const int h = idx & 15, lane = (idx >> 4) & 31, frag = idx >> 9;
  const int ks = frag % 12, mtile = (frag / 12) & 7, sg = frag / 96;
  const int m = mtile * 16 + (lane & 15);
  const int k = 32 * ks + (lane >> 4) * 16 + h;  // < 384
  const int dt = k >> 7, c = k & 127;
  wrep2[idx] = (_Float16)w.p[sg][(m * HID_ + c) * 3 + dt];
}

// ---------------------------------------------------------------------------
// 3) Fused encoder: one workgroup = (sig, b*n series, 64-wide T chunk).
//    8 waves; wave w owns output-channel tile [16w, 16w+16).
// ---------------------------------------------------------------------------
__global__ __launch_bounds__(256)
void encoder_kernel(const _Float16* __restrict__ sig,
                    const _Float16* __restrict__ wrep1,
                    const _Float16* __restrict__ wrep2,
                    EncArgs ea,
                    float* __restrict__ partial,   // [3][BN_][NCH_][128]
                    float* __restrict__ hlast) {   // [3][B][N][12][128]
  const int chunk = blockIdx.x;            // 0..8
  const int bn    = blockIdx.y;            // 0..799
  const int s     = blockIdx.z;            // 0..2
  const int tb    = chunk * 64;
  const int tid = threadIdx.x, lane = tid & 31, wave = tid >> 5;
  const int lo = lane & 15, hi = lane >> 4;
  const int mtile = wave;

  __shared__ alignas(32) _Float16 sigRow[CIN_][68];   // sig t in [tb-2, tb+66)
  __shared__ alignas(32) _Float16 B1[80][64];         // conv1 im2col (zero-padded)
  __shared__ alignas(32) _Float16 h1T[66][HID_];      // conv1 out, [t][c], t=tb-1+j
  __shared__ float b1s[HID_], b2s[HID_];

  for (int i = tid; i < HID_; i += 256) { b1s[i] = ea.b1[s][i]; b2s[i] = ea.b2[s][i]; }

  const _Float16* sb = sig + ((size_t)(s * BN_ + bn) * CIN_) * T_;
  for (int i = tid; i < CIN_ * 68; i += 256) {
    int c = i / 68, tt = i % 68;
    int t = tb - 2 + tt;
    sigRow[c][tt] = (t >= 0 && t < T_) ? sb[(size_t)c * T_ + t] : (_Float16)0.f;
  }
  __syncthreads();
  for (int i = tid; i < 80 * 64; i += 256) {
    int j = i >> 6, k = i & 63;
    _Float16 v = (_Float16)0.f;
    if (j < 66 && k < 42) { int dt = k / CIN_, c = k % CIN_; v = sigRow[c][j + dt]; }
    B1[j][k] = v;
  }
  __syncthreads();

  // ---- conv1: [128 x 42] * [42 x 66] -> GELU -> h1T (transposed, f16) ----
  const v16h a0 = *(const v16h*)(wrep1 + (((size_t)(s * 8 + mtile) * 2 + 0) * 32 + lane) * 16);
  const v16h a1 = *(const v16h*)(wrep1 + (((size_t)(s * 8 + mtile) * 2 + 1) * 32 + lane) * 16);
#pragma unroll
  for (int nt = 0; nt < 5; ++nt) {
    const int j = nt * 16 + lo;                  // 0..79
    const v16h bfr0 = *(const v16h*)&B1[j][hi * 16];
    const v16h bfr1 = *(const v16h*)&B1[j][32 + hi * 16];
    v8f acc = {};
    acc = wmma_f16(a0, bfr0, acc);
    acc = wmma_f16(a1, bfr1, acc);
    if (j < 66) {
      const int t = tb - 1 + j;                  // conv2 zero-pads h1 at t=-1,T
      const float live = (t >= 0 && t < T_) ? 1.0f : 0.0f;
      v8h outv;
#pragma unroll
      for (int r = 0; r < 8; ++r) {
        const int o = mtile * 16 + hi * 8 + r;
        const float v = live * gelu_f(acc[r] + b1s[o]);
        outv[r] = (_Float16)v;
      }
      *(v8h*)&h1T[j][mtile * 16 + hi * 8] = outv;   // 16B aligned ds_store
    }
  }
  __syncthreads();

  // ---- conv2: [128 x 384] * [384 x 64], K split in 12 steps of 32 ----
  v8f acc2[4] = {v8f{}, v8f{}, v8f{}, v8f{}};
#pragma unroll
  for (int ks = 0; ks < 12; ++ks) {
    const v16h a = *(const v16h*)(wrep2 + (((size_t)(s * 8 + mtile) * 12 + ks) * 32 + lane) * 16);
    const int kb = 32 * ks + hi * 16;            // k = dt*128 + c
    const int dt = kb >> 7, c0 = kb & 127;
#pragma unroll
    for (int nt = 0; nt < 4; ++nt) {
      const int j2 = nt * 16 + lo;               // t = tb + j2
      const v16h bb = *(const v16h*)&h1T[j2 + dt][c0];   // 32B aligned ds_load
      acc2[nt] = wmma_f16(a, bb, acc2[nt]);
    }
  }

  // ---- epilogue: bias + GELU; last-12 slice; channel sums ----
  const int b = bn / N_, n = bn % N_;
  float osum[8] = {0.f, 0.f, 0.f, 0.f, 0.f, 0.f, 0.f, 0.f};
#pragma unroll
  for (int nt = 0; nt < 4; ++nt) {
    const int t = tb + nt * 16 + lo;
#pragma unroll
    for (int r = 0; r < 8; ++r) {
      const int o = mtile * 16 + hi * 8 + r;
      const float v = gelu_f(acc2[nt][r] + b2s[o]);
      osum[r] += v;
      if (t >= LAST0)
        hlast[((((size_t)s * B_ + b) * N_ + n) * HZ_ + (t - LAST0)) * HID_ + o] = v;
    }
  }
#pragma unroll
  for (int r = 0; r < 8; ++r) {
    float v = osum[r];
    v += __shfl_xor(v, 8); v += __shfl_xor(v, 4);
    v += __shfl_xor(v, 2); v += __shfl_xor(v, 1);
    if (lo == 0) {
      const int o = mtile * 16 + hi * 8 + r;
      partial[(((size_t)(s * BN_ + bn)) * NCH_ + chunk) * HID_ + o] = v;
    }
  }
}

// ---------------------------------------------------------------------------
// 4) deterministic mean partial reduction -> meanAcc[(sg*8+b)*128 + o]
// ---------------------------------------------------------------------------
__global__ void reduce_mean_kernel(const float* __restrict__ partial,
                                   float* __restrict__ meanAcc) {
  const int idx = blockIdx.x * 256 + threadIdx.x;  // 3*8*128 = 3072
  if (idx >= 3 * B_ * HID_) return;
  const int o = idx & 127, b = (idx >> 7) & 7, sg = idx >> 10;
  float sum = 0.f;
  for (int n = 0; n < N_; ++n)
    for (int ch = 0; ch < NCH_; ++ch)
      sum += partial[(((size_t)(sg * BN_ + b * N_ + n)) * NCH_ + ch) * HID_ + o];
  meanAcc[idx] = sum;
}

// ---------------------------------------------------------------------------
// 5) head: predictor heads + gate softmax + blend, per (b,n,t12)
// ---------------------------------------------------------------------------
__global__ __launch_bounds__(128)
void head_kernel(const float* __restrict__ hlast, HeadArgs ha,
                 float* __restrict__ outPred) {
  const int blk = blockIdx.x;                 // b*1200 + n*12 + t12
  const int t12 = blk % HZ_, n = (blk / HZ_) % N_, b = blk / (N_ * HZ_);
  const int tid = threadIdx.x;                // 128
  __shared__ float hc[3 * HID_], ug[HID_], vp[3][64], gy[8];

  for (int i = tid; i < 3 * HID_; i += 128) {
    const int sg = i >> 7, o = i & 127;
    hc[i] = hlast[((((size_t)sg * B_ + b) * N_ + n) * HZ_ + t12) * HID_ + o];
  }
  __syncthreads();
  {
    float a = ha.gb1[tid];
    for (int k = 0; k < 3 * HID_; ++k) a += ha.gw1[(size_t)tid * 384 + k] * hc[k];
    ug[tid] = gelu_f(a);
  }
  if (tid < 64) {
#pragma unroll
    for (int sg = 0; sg < 3; ++sg) {
      float a = ha.pb1[sg][tid];
      for (int k = 0; k < HID_; ++k) a += ha.pw1[sg][tid * HID_ + k] * hc[sg * HID_ + k];
      vp[sg][tid] = gelu_f(a);
    }
  }
  __syncthreads();
  if (tid < 3) {
    float g = ha.gb2[tid];
    for (int k = 0; k < HID_; ++k) g += ha.gw2[tid * HID_ + k] * ug[k];
    gy[tid] = g;
    float y = ha.pb2[tid][0];
    for (int k = 0; k < 64; ++k) y += ha.pw2[tid][k] * vp[tid][k];
    gy[4 + tid] = y;
  }
  __syncthreads();
  if (tid == 0) {
    const float m = fmaxf(gy[0], fmaxf(gy[1], gy[2]));
    const float e0 = __expf(gy[0] - m), e1 = __expf(gy[1] - m), e2 = __expf(gy[2] - m);
    const float inv = __builtin_amdgcn_rcpf(e0 + e1 + e2);
    outPred[((size_t)b * N_ + n) * HZ_ + t12] = (e0 * gy[4] + e1 * gy[5] + e2 * gy[6]) * inv;
  }
}

// ---------------------------------------------------------------------------
// 6) projection MLP + L2 normalize -> rep[B,64]
// ---------------------------------------------------------------------------
__global__ __launch_bounds__(384)
void proj_kernel(const float* __restrict__ meanAcc,
                 const float* __restrict__ w1, const float* __restrict__ b1,
                 const float* __restrict__ w2, const float* __restrict__ b2,
                 float* __restrict__ outRep) {
  const int b = blockIdx.x, tid = threadIdx.x;   // 384 threads
  __shared__ float z[384], hid[384], zo[64], nrm;
  {
    const int sg = tid >> 7, o = tid & 127;
    z[tid] = meanAcc[((size_t)sg * B_ + b) * HID_ + o] * (1.0f / (N_ * T_));
  }
  __syncthreads();
  {
    float a = b1[tid];
    for (int k = 0; k < 384; ++k) a += w1[(size_t)tid * 384 + k] * z[k];
    hid[tid] = gelu_f(a);
  }
  __syncthreads();
  if (tid < 64) {
    float a = b2[tid];
    for (int k = 0; k < 384; ++k) a += w2[(size_t)tid * 384 + k] * hid[k];
    zo[tid] = a;
  }
  __syncthreads();
  if (tid == 0) {
    float s = 0.f;
    for (int k = 0; k < 64; ++k) s += zo[k] * zo[k];
    nrm = fmaxf(sqrtf(s), 1e-12f);
  }
  __syncthreads();
  if (tid < 64) outRep[(size_t)b * 64 + tid] = zo[tid] / nrm;
}

// ---------------------------------------------------------------------------
// host launcher
// ---------------------------------------------------------------------------
static inline size_t align256(size_t x) { return (x + 255) & ~(size_t)255; }

extern "C" void kernel_launch(void* const* d_in, const int* in_sizes, int n_in,
                              void* d_out, int out_size, void* d_ws, size_t ws_size,
                              hipStream_t stream) {
  (void)in_sizes; (void)n_in; (void)out_size; (void)ws_size;
  // d_in layout (setup_inputs insertion order, leaves depth-first):
  // 0:x; enc(hf=1,daily=5,trend=9):{w1,b1,w2,b2}; pred(hf=13,daily=17,trend=21):
  // {w1,b1,w2,b2}; gate:25..28 {w1,b1,w2,b2}; proj:29..32 {w1,b1,w2,b2}
  const float* x = (const float*)d_in[0];
  const int encB[3]  = {1, 5, 9};
  const int predB[3] = {13, 17, 21};

  W3 w1s, w2s; EncArgs ea; HeadArgs ha;
  for (int s = 0; s < 3; ++s) {
    w1s.p[s]  = (const float*)d_in[encB[s] + 0];
    ea.b1[s]  = (const float*)d_in[encB[s] + 1];
    w2s.p[s]  = (const float*)d_in[encB[s] + 2];
    ea.b2[s]  = (const float*)d_in[encB[s] + 3];
    ha.pw1[s] = (const float*)d_in[predB[s] + 0];
    ha.pb1[s] = (const float*)d_in[predB[s] + 1];
    ha.pw2[s] = (const float*)d_in[predB[s] + 2];
    ha.pb2[s] = (const float*)d_in[predB[s] + 3];
  }
  ha.gw1 = (const float*)d_in[25]; ha.gb1 = (const float*)d_in[26];
  ha.gw2 = (const float*)d_in[27]; ha.gb2 = (const float*)d_in[28];
  const float* pjw1 = (const float*)d_in[29];
  const float* pjb1 = (const float*)d_in[30];
  const float* pjw2 = (const float*)d_in[31];
  const float* pjb2 = (const float*)d_in[32];

  // workspace carve (all 256B aligned)
  char* ws = (char*)d_ws;
  size_t off = 0;
  _Float16* sig   = (_Float16*)(ws + off); off += align256((size_t)3 * BN_ * CIN_ * T_ * 2);
  _Float16* wrep1 = (_Float16*)(ws + off); off += align256((size_t)3 * 8 * 2 * 512 * 2);
  _Float16* wrep2 = (_Float16*)(ws + off); off += align256((size_t)3 * 8 * 12 * 512 * 2);
  float* partial  = (float*)(ws + off);    off += align256((size_t)3 * BN_ * NCH_ * HID_ * 4);
  float* meanAcc  = (float*)(ws + off);    off += align256((size_t)3 * B_ * HID_ * 4);
  float* hlast    = (float*)(ws + off);    off += align256((size_t)3 * B_ * N_ * HZ_ * HID_ * 4);

  float* outPred = (float*)d_out;                      // [8,1,100,12] = 9600
  float* outRep  = (float*)d_out + (size_t)B_ * N_ * HZ_;  // [8,64]

  decompose_kernel<<<dim3(CIN_, N_, B_), T_, 0, stream>>>(x, sig);
  repack_w1_kernel<<<96, 256, 0, stream>>>(w1s, wrep1);
  repack_w2_kernel<<<576, 256, 0, stream>>>(w2s, wrep2);
  encoder_kernel<<<dim3(NCH_, BN_, 3), 256, 0, stream>>>(sig, wrep1, wrep2, ea,
                                                         partial, hlast);
  reduce_mean_kernel<<<12, 256, 0, stream>>>(partial, meanAcc);
  head_kernel<<<B_ * N_ * HZ_, 128, 0, stream>>>(hlast, ha, outPred);
  proj_kernel<<<B_, 384, 0, stream>>>(meanAcc, pjw1, pjb1, pjw2, pjb2, outRep);
}
